// LPropDecoderLayer_60370060312791
// MI455X (gfx1250) — matrix operand order repaired and checked by reading, not compiled, not run on hardware
//
#include <hip/hip_runtime.h>
#include <hip/hip_bf16.h>

// ---------------------------------------------------------------------------
// CDNA5 (gfx1250) implementation: bf16 WMMA GEMMs + flash attention.
// WMMA fragment layouts per CDNA5 ISA 7.12.2 (wave32):
//   A 16x32 bf16 : lane L (m=L%16,h=L/16), elem i -> k = (i<8?0:16)+8h+(i&7)
//   B 32x16 bf16 : lane L (col=L%16,h=L/16), elem i -> k = (i<8?0:16)+8h+(i&7)  (as A)
//   C/D 16x16 f32: VGPR r, lane L -> row M = r+8h, col N = L%16
// ---------------------------------------------------------------------------

typedef __attribute__((ext_vector_type(16))) __bf16 bf16x16;
typedef __attribute__((ext_vector_type(8)))  __bf16 bf16x8;
typedef __attribute__((ext_vector_type(8)))  float  f32x8;
typedef __attribute__((ext_vector_type(4)))  float  f32x4;

#define TAU_INV 30.0f

static __device__ __forceinline__ f32x8 wmma_bf16(bf16x16 a, bf16x16 b, f32x8 c) {
  // D = A*B + C   (8-arg form: neg_a, A, neg_b, B, c_mod, C, reuse_a, reuse_b)
  return __builtin_amdgcn_wmma_f32_16x16x32_bf16(false, a, false, b, (short)0, c,
                                                 false, false);
}

static __device__ __forceinline__ bf16x16 cat8(bf16x8 lo, bf16x8 hi) {
  return __builtin_shufflevector(lo, hi, 0,1,2,3,4,5,6,7,8,9,10,11,12,13,14,15);
}

// ---------------------------------------------------------------------------
// Projection GEMM:  out[n][dm][s] = sum_ch W[dm][ch] * X(n,ch,s) + bias[dm]
// X addressing: X + n*nStride + (s>>10)*fStride + ch*1024 + (s&1023)
// One wave per 16x16 output tile; K = 512 in chunks of 32.
// ---------------------------------------------------------------------------
__global__ __launch_bounds__(32)
void proj_gemm_kernel(const float* __restrict__ X, const float* __restrict__ W,
                      const float* __restrict__ bias, float* __restrict__ out,
                      int S, int sTiles,
                      unsigned long long nStride, unsigned long long fStride)
{
  __shared__ __bf16 ldsB[16 * 32];              // [s_local][ch_local] bf16
  const int lane = threadIdx.x;
  const int m = lane & 15, h = lane >> 4;
  const int perN = 8 * sTiles;
  const int n   = blockIdx.x / perN;
  const int rem = blockIdx.x - n * perN;
  const int dm0 = (rem / sTiles) * 16;
  const int s0  = (rem % sTiles) * 16;
  const int f   = s0 >> 10;
  const int j0  = s0 & 1023;
  const float* Xn = X + (unsigned long long)n * nStride
                      + (unsigned long long)f * fStride + j0;

  f32x8 acc = {0.f,0.f,0.f,0.f,0.f,0.f,0.f,0.f};

  for (int k0 = 0; k0 < 512; k0 += 32) {
    // ---- A fragment from W (rows contiguous in ch) ----
    const float* wr = W + (unsigned long long)(dm0 + m) * 512 + k0 + 8 * h;
    f32x4 w0 = *(const f32x4*)(wr);
    f32x4 w1 = *(const f32x4*)(wr + 4);
    f32x4 w2 = *(const f32x4*)(wr + 16);
    f32x4 w3 = *(const f32x4*)(wr + 20);
    bf16x16 a;
#pragma unroll
    for (int i = 0; i < 4; ++i) {
      a[i]      = (__bf16)w0[i];  a[4 + i]  = (__bf16)w1[i];
      a[8 + i]  = (__bf16)w2[i];  a[12 + i] = (__bf16)w3[i];
    }
    // ---- Stage B tile: X rows are s-contiguous; LDS holds [s][ch] ----
    const float* xr = Xn + (unsigned long long)(k0 + lane) * 1024;
    f32x4 x0 = *(const f32x4*)(xr);
    f32x4 x1 = *(const f32x4*)(xr + 4);
    f32x4 x2 = *(const f32x4*)(xr + 8);
    f32x4 x3 = *(const f32x4*)(xr + 12);
#pragma unroll
    for (int i = 0; i < 4; ++i) {
      ldsB[(0  + i) * 32 + lane] = (__bf16)x0[i];
      ldsB[(4  + i) * 32 + lane] = (__bf16)x1[i];
      ldsB[(8  + i) * 32 + lane] = (__bf16)x2[i];
      ldsB[(12 + i) * 32 + lane] = (__bf16)x3[i];
    }
    // Single wave: DS ops are in-order, no barrier needed.
    bf16x16 b = cat8(*(const bf16x8*)(&ldsB[m * 32 + 8 * h]), *(const bf16x8*)(&ldsB[m * 32 + 16 + 8 * h])); // table-order ch per lane
    acc = wmma_bf16(a, b, acc);
  }

#pragma unroll
  for (int r = 0; r < 8; ++r) {
    const int dm = dm0 + 8 * h + r;
    out[((unsigned long long)n * 128 + dm) * (unsigned long long)S + s0 + m] =
        acc[r] + bias[dm];
  }
}

// ---------------------------------------------------------------------------
// L2-normalize (over d=128) and repack to bf16 [n][s][128] rows.
// This single layout serves BOTH the A(Q) and B(K^T) score-GEMM fragments.
// ---------------------------------------------------------------------------
__global__ __launch_bounds__(256)
void l2norm_pack_kernel(const float* __restrict__ in, __bf16* __restrict__ outp, int S)
{
  const int id = blockIdx.x * 256 + threadIdx.x;
  const int n = id / S, s = id - n * S;
  const float* src = in + (unsigned long long)n * 128ull * S + s;
  float sum = 0.f;
#pragma unroll 8
  for (int d = 0; d < 128; ++d) {
    float x = src[(unsigned long long)d * S];
    sum += x * x;
  }
  const float rn = 1.0f / fmaxf(sqrtf(sum), 1e-12f);
  __bf16* dst = outp + ((unsigned long long)n * S + s) * 128ull;
#pragma unroll 8
  for (int d = 0; d < 128; ++d)
    dst[d] = (__bf16)(src[(unsigned long long)d * S] * rn);
}

// ---------------------------------------------------------------------------
// Residual + InstanceNorm over hw=1024 per (n, ch). One block per (n,ch).
// ---------------------------------------------------------------------------
__global__ __launch_bounds__(256)
void instnorm_kernel(const float* __restrict__ tgt, const float* __restrict__ attn,
                     float* __restrict__ outp)
{
  __shared__ float s1[256], s2[256];
  const unsigned long long nc = blockIdx.x;
  const float* a = tgt  + nc * 1024ull;
  const float* b = attn + nc * 1024ull;
  const int t = threadIdx.x;
  float x[4], sum = 0.f, sq = 0.f;
#pragma unroll
  for (int i = 0; i < 4; ++i) {
    x[i] = a[t + 256 * i] + b[t + 256 * i];
    sum += x[i]; sq += x[i] * x[i];
  }
  s1[t] = sum; s2[t] = sq;
  __syncthreads();
  for (int o = 128; o > 0; o >>= 1) {
    if (t < o) { s1[t] += s1[t + o]; s2[t] += s2[t + o]; }
    __syncthreads();
  }
  const float mu  = s1[0] * (1.f / 1024.f);
  const float var = s2[0] * (1.f / 1024.f) - mu * mu;
  const float rs  = rsqrtf(var + 1e-5f);
  float* o = outp + nc * 1024ull;
#pragma unroll
  for (int i = 0; i < 4; ++i) o[t + 256 * i] = (x[i] - mu) * rs;
}

// ---------------------------------------------------------------------------
// Flash attention: one wave handles a [16 q x 128 ch] output slab; 4 waves per
// block cover ch=512 (scores recomputed per wave: 60% FLOP overhead, keeps the
// O accumulator at 64 VGPRs/lane). Online softmax with shfl_xor reductions
// inside each 16-lane half (C-fragment rows M=r+8h live per half).
// V is read directly from the raw f32 tensor ([ch][k] contiguity == B frag).
// ---------------------------------------------------------------------------
__global__ __launch_bounds__(128)
void flash_attn_kernel(const __bf16* __restrict__ Qb, const __bf16* __restrict__ Kb,
                       const float* __restrict__ V, float* __restrict__ outp,
                       int Sq, int Sk, int qTiles,
                       unsigned long long vNStride, unsigned long long vFStride,
                       int doSigmoid)
{
  __shared__ __bf16 pbuf[4][16 * 32];           // per-wave P tile [q][k]
  const int lane = threadIdx.x & 31;
  const int wv   = threadIdx.x >> 5;
  const int m = lane & 15, h = lane >> 4;
  const int n   = blockIdx.x / qTiles;
  const int q0  = (blockIdx.x - n * qTiles) * 16;
  const int ch0 = wv * 128;
  __bf16* pl = pbuf[wv];

  // Q A-fragments (d = 128 -> 4 chunks of 32)
  const __bf16* qrow = Qb + ((unsigned long long)n * Sq + q0 + m) * 128ull;
  bf16x16 aQ[4];
#pragma unroll
  for (int j = 0; j < 4; ++j) {
    bf16x8 lo = *(const bf16x8*)(qrow + 32 * j + 16 * h);
    bf16x8 hi = *(const bf16x8*)(qrow + 32 * j + 16 * h + 8);
    aQ[j] = cat8(lo, hi);
  }

  f32x8 O[8];
#pragma unroll
  for (int t = 0; t < 8; ++t) O[t] = (f32x8){0.f,0.f,0.f,0.f,0.f,0.f,0.f,0.f};
  float mrow[8], lrow[8];
#pragma unroll
  for (int r = 0; r < 8; ++r) { mrow[r] = -1e30f; lrow[r] = 0.f; }

  const unsigned long long kRowBase = (unsigned long long)n * Sk;
  const float* Vn = V + (unsigned long long)n * vNStride;

  for (int k0 = 0; k0 < Sk; k0 += 32) {
    // ---- scores for two 16-column chunks ----
    f32x8 c0 = (f32x8){0.f,0.f,0.f,0.f,0.f,0.f,0.f,0.f};
    f32x8 c1 = c0;
#pragma unroll
    for (int j = 0; j < 4; ++j) {
      bf16x16 bK = *(const bf16x16*)(Kb + (kRowBase + k0 + m) * 128ull + 32 * j + 16 * h);
      c0 = wmma_bf16(aQ[j], bK, c0);
    }
#pragma unroll
    for (int j = 0; j < 4; ++j) {
      bf16x16 bK = *(const bf16x16*)(Kb + (kRowBase + k0 + 16 + m) * 128ull + 32 * j + 16 * h);
      c1 = wmma_bf16(aQ[j], bK, c1);
    }
    // ---- online softmax (per row q = q0 + 8h + r) ----
    float alpha[8];
#pragma unroll
    for (int r = 0; r < 8; ++r) {
      float x0 = c0[r] * TAU_INV;
      float x1 = c1[r] * TAU_INV;
      float mx = fmaxf(x0, x1);
      mx = fmaxf(mx, __shfl_xor(mx, 1, 16));
      mx = fmaxf(mx, __shfl_xor(mx, 2, 16));
      mx = fmaxf(mx, __shfl_xor(mx, 4, 16));
      mx = fmaxf(mx, __shfl_xor(mx, 8, 16));
      const float mnew = fmaxf(mrow[r], mx);
      const float al = __expf(mrow[r] - mnew);
      const float p0 = __expf(x0 - mnew);
      const float p1 = __expf(x1 - mnew);
      float ps = p0 + p1;
      ps += __shfl_xor(ps, 1, 16);
      ps += __shfl_xor(ps, 2, 16);
      ps += __shfl_xor(ps, 4, 16);
      ps += __shfl_xor(ps, 8, 16);
      lrow[r] = lrow[r] * al + ps;
      mrow[r] = mnew;
      alpha[r] = al;
      pl[(8 * h + r) * 32 + m]      = (__bf16)p0;   // P tile [q][k] in LDS
      pl[(8 * h + r) * 32 + 16 + m] = (__bf16)p1;
    }
    // ---- rescale accumulator ----
#pragma unroll
    for (int t = 0; t < 8; ++t)
#pragma unroll
      for (int r = 0; r < 8; ++r) O[t][r] *= alpha[r];
    // ---- P A-fragment via LDS transpose (in-wave DS is in-order) ----
    bf16x8 plo = *(const bf16x8*)(pl + m * 32 + 16 * h);
    bf16x8 phi = *(const bf16x8*)(pl + m * 32 + 16 * h + 8);
    bf16x16 aP = cat8(plo, phi);
    // ---- V B-fragments straight from global f32, accumulate ----
    const int kg = k0 + 16 * h;
    const float* vb = Vn + (unsigned long long)(kg >> 10) * vFStride + (kg & 1023);
#pragma unroll
    for (int t = 0; t < 8; ++t) {
      const float* vp = vb + (unsigned long long)(ch0 + 16 * t + m) * 1024ull;
      f32x4 v0 = *(const f32x4*)(vp);
      f32x4 v1 = *(const f32x4*)(vp + 4);
      f32x4 v2 = *(const f32x4*)(vp + 8);
      f32x4 v3 = *(const f32x4*)(vp + 12);
      bf16x16 bV;
#pragma unroll
      for (int i = 0; i < 4; ++i) {
        bV[i]      = (__bf16)v0[i];  bV[4 + i]  = (__bf16)v1[i];
        bV[8 + i]  = (__bf16)v2[i];  bV[12 + i] = (__bf16)v3[i];
      }
      O[t] = wmma_bf16(aP, bV, O[t]);
    }
  }

  // ---- epilogue: normalize, optional sigmoid, store [n][ch][q] ----
#pragma unroll
  for (int t = 0; t < 8; ++t) {
    f32x4 lo4, hi4;
#pragma unroll
    for (int r = 0; r < 8; ++r) {
      float v = O[t][r] / lrow[r];
      if (doSigmoid) v = 1.0f / (1.0f + __expf(-v));
      if (r < 4) lo4[r] = v; else hi4[r - 4] = v;
    }
    float* op = outp + ((unsigned long long)n * 512 + ch0 + 16 * t + m)
                         * (unsigned long long)Sq + q0 + 8 * h;
    *(f32x4*)(op)     = lo4;
    *(f32x4*)(op + 4) = hi4;
  }
}

// ---------------------------------------------------------------------------
// Host-side orchestration (graph-capture safe: launches only).
// ---------------------------------------------------------------------------
extern "C" void kernel_launch(void* const* d_in, const int* in_sizes, int n_in,
                              void* d_out, int out_size, void* d_ws, size_t ws_size,
                              hipStream_t stream)
{
  (void)in_sizes; (void)n_in; (void)out_size; (void)ws_size;
  const float* tgt    = (const float*)d_in[0];   // [1,4,512,32,32]
  const float* memory = (const float*)d_in[1];   // [8,4,512,32,32]
  const float* pos    = (const float*)d_in[2];   // [8,4,512,32,32]
  const float* Wself  = (const float*)d_in[3];   // [128,512]
  const float* bself  = (const float*)d_in[4];   // [128]
  const float* Wcross = (const float*)d_in[5];   // [128,512]
  const float* bcross = (const float*)d_in[6];   // [128]
  float* out = (float*)d_out;                    // [1,4,512,32,32]

  char* ws = (char*)d_ws;                        // ~42 MB used
  float*  Rproj  = (float*) (ws);                // 16 MB  f32 proj scratch (reused 3x)
  __bf16* RselfW = (__bf16*)(ws + (16ull << 20));// 1 MB   self wq/wk bf16 [n][1024][128]
  float*  RattnO = (float*) (ws + (17ull << 20));// 8 MB   self attn out [n][512][1024]
  float*  Rtgtn  = (float*) (ws + (25ull << 20));// 8 MB   normalized tgt [n][512][1024]
  __bf16* RQc    = (__bf16*)(ws + (33ull << 20));// 1 MB   cross Q bf16
  __bf16* RKc    = (__bf16*)(ws + (34ull << 20));// 8 MB   cross K bf16 [n][8192][128]

  const unsigned long long CHW = 512ull * 1024ull;  // batch stride (elements)
  const unsigned long long FS  = 4ull * CHW;        // frame stride for memory/pos

  // 1) self projection + l2norm
  proj_gemm_kernel<<<4 * 8 * 64, 32, 0, stream>>>(tgt, Wself, bself, Rproj,
                                                  1024, 64, CHW, 0ull);
  l2norm_pack_kernel<<<(4 * 1024) / 256, 256, 0, stream>>>(Rproj, RselfW, 1024);
  // 2) self flash attention (V = tgt), out -> RattnO
  flash_attn_kernel<<<4 * 64, 128, 0, stream>>>(RselfW, RselfW, tgt, RattnO,
                                                1024, 1024, 64, CHW, 0ull, 0);
  // 3) residual + instance norm
  instnorm_kernel<<<4 * 512, 256, 0, stream>>>(tgt, RattnO, Rtgtn);
  // 4) cross Q projection + l2norm
  proj_gemm_kernel<<<4 * 8 * 64, 32, 0, stream>>>(Rtgtn, Wcross, bcross, Rproj,
                                                  1024, 64, CHW, 0ull);
  l2norm_pack_kernel<<<(4 * 1024) / 256, 256, 0, stream>>>(Rproj, RQc, 1024);
  // 5) cross K projection (memory) + l2norm
  proj_gemm_kernel<<<4 * 8 * 512, 32, 0, stream>>>(memory, Wcross, bcross, Rproj,
                                                   8192, 512, CHW, FS);
  l2norm_pack_kernel<<<(4 * 8192) / 256, 256, 0, stream>>>(Rproj, RKc, 8192);
  // 6) cross flash attention (V = pos_enc) + sigmoid -> d_out
  flash_attn_kernel<<<4 * 64, 128, 0, stream>>>(RQc, RKc, pos, out,
                                                1024, 8192, 64, CHW, FS, 1);
}
